// EDGSDE_7653631721659
// MI455X (gfx1250) — matrix operand order repaired and checked by reading, not compiled, not run on hardware
//
#include <hip/hip_runtime.h>
#include <hip/hip_bf16.h>
#include <math.h>

typedef unsigned short u16;
typedef __attribute__((ext_vector_type(16))) __bf16 v16bf;
typedef __attribute__((ext_vector_type(8)))  float  v8f;

union Frag {
    v16bf v;
    uint4 q[2];
};

// ---------- constants ----------
#define T_    8
#define NSUB_ 4
#define K_    64
#define NB_   16
#define D_    512
#define H_    512
#define B_    512
#define STEPS_ 64
#define DT_    0.015625f      /* 1/64  */
#define SQDT_  0.125f         /* 1/8   */
#define EPS_   1e-7f

// workspace layout (bytes)
#define WT_BYTES    ((size_t)4 * 64 * 512 * 512 * 2)        /* 128 MB packed bf16 weights  */
#define PATH_BYTES  ((size_t)8 * 64 * 16 * 512 * 4)         /* 16 MB saved path f32        */
#define ZB_BYTES    ((size_t)8 * 512 * 512 * 2)             /* 4 MB z bf16                 */
#define ZHB_BYTES   ((size_t)8 * 1024 * 512 * 2)            /* 8 MB zhat bf16              */
#define G_BYTES     ((size_t)8 * 1024 * 512 * 4)            /* 16 MB similarity matrix     */
#define ZN_BYTES    ((size_t)8 * 512 * 4)                   /* z row norms                 */
#define ACC_BYTES   ((size_t)16 * 4)                        /* mse[8], ce[8]               */

// ---------- helpers ----------
__device__ __forceinline__ u16 f32_to_bf16(float f) {
    unsigned u = __float_as_uint(f);
    unsigned r = (u + 0x7FFFu + ((u >> 16) & 1u)) >> 16;
    return (u16)r;
}

__device__ __forceinline__ v8f vzero8() {
    v8f z;
#pragma unroll
    for (int i = 0; i < 8; ++i) z[i] = 0.0f;
    return z;
}

// A-fragment (16x32 bf16): lanes 0-15 hold M rows, slots 0..7 = K(dbase+off..),
// slots 8..15 = K(dbase+16+off..), off = 8*(lane>=16)
__device__ __forceinline__ v16bf loadA16(const u16* rowp, int dbase, int lane) {
    Frag f;
    int off = ((lane >> 4) << 3);
    f.q[0] = *reinterpret_cast<const uint4*>(rowp + dbase + off);
    f.q[1] = *reinterpret_cast<const uint4*>(rowp + dbase + 16 + off);
    return f.v;
}

// B-fragment (32x16 bf16): lanes 0-15 hold N cols, slots 0..7 = K(dbase+off..),
// slots 8..15 = K(dbase+16+off..), off = 8*(lane>=16)
__device__ __forceinline__ v16bf loadB16(const u16* rowp, int dbase, int lane) {
    Frag f;
    int off = ((lane >> 4) << 3);
    f.q[0] = *reinterpret_cast<const uint4*>(rowp + dbase + off);
    f.q[1] = *reinterpret_cast<const uint4*>(rowp + dbase + 16 + off);
    return f.v;
}

// one wave computes 4 adjacent 16x16 output tiles of (16x512)@(512x512)
__device__ __forceinline__ void wave_gemm4(const u16* __restrict__ Alds,
                                           const u16* __restrict__ Bglob,
                                           int wv, int lane, v8f acc[4]) {
    const u16* arowp = Alds + (size_t)(lane & 15) * D_;
#pragma unroll 4
    for (int kt = 0; kt < 16; ++kt) {
        const int dbase = kt * 32;
        v16bf a = loadA16(arowp, dbase, lane);
#pragma unroll
        for (int nt = 0; nt < 4; ++nt) {
            const int h = ((wv * 4 + nt) * 16) + (lane & 15);
            v16bf b = loadB16(Bglob + (size_t)h * D_, dbase, lane);
            acc[nt] = __builtin_amdgcn_wmma_f32_16x16x32_bf16(
                false, a, false, b, (short)0, acc[nt], false, false);
        }
    }
}

__device__ __forceinline__ void act_tanh_store(const v8f acc[4],
                                               const float* __restrict__ biasRow,
                                               u16* __restrict__ outLds,
                                               int wv, int lane) {
#pragma unroll
    for (int nt = 0; nt < 4; ++nt) {
        int col = ((wv * 4 + nt) * 16) + (lane & 15);
        float b = biasRow[col];
#pragma unroll
        for (int v = 0; v < 8; ++v) {
            int m = v + ((lane >> 4) << 3);
            outLds[m * D_ + col] = f32_to_bf16(tanhf(acc[nt][v] + b));
        }
    }
}

// ---------- kernel 1: pack+transpose weights f32 -> bf16 [w][k][h][d] ----------
__global__ __launch_bounds__(256) void pack_kernel(const float* __restrict__ Wf1,
                                                   const float* __restrict__ Wf2,
                                                   const float* __restrict__ Wg1,
                                                   const float* __restrict__ Wg2,
                                                   u16* __restrict__ Wt) {
    __shared__ float tile[32][33];
    int bid = blockIdx.x;
    int tx = bid & 15; bid >>= 4;
    int ty = bid & 15; bid >>= 4;
    int k  = bid & 63; bid >>= 6;
    int w  = bid; // 0..3
    const float* src = (w == 0) ? Wf1 : (w == 1) ? Wf2 : (w == 2) ? Wg1 : Wg2;
    src += (size_t)k * (D_ * H_);
    u16* dst = Wt + ((size_t)w * 64 + k) * (D_ * H_);
    int c = threadIdx.x & 31;
    int r0 = threadIdx.x >> 5;
#pragma unroll
    for (int rr = 0; rr < 4; ++rr) {
        int r = r0 + rr * 8;
        tile[r][c] = src[(size_t)(ty * 32 + r) * 512 + tx * 32 + c];
    }
    __syncthreads();
#pragma unroll
    for (int rr = 0; rr < 4; ++rr) {
        int h = tx * 32 + r0 + rr * 8;
        int d = ty * 32 + c;
        dst[(size_t)h * 512 + d] = f32_to_bf16(tile[c][r0 + rr * 8]);
    }
}

// ---------- kernel 2: the SDE chain, one workgroup per k ----------
__global__ __launch_bounds__(256, 1) void sde_kernel(
    const float* __restrict__ init_mu, const float* __restrict__ init_noise,
    const float* __restrict__ dW,
    const float* __restrict__ bf1, const float* __restrict__ bf2,
    const float* __restrict__ bg1, const float* __restrict__ bg2,
    const u16* __restrict__ Wt, float* __restrict__ pathBuf) {
    __shared__ float yF[NB_ * D_];   // 32 KB, f32 state
    __shared__ u16  yB[NB_ * D_];    // 16 KB, bf16 state (A-operand source)
    __shared__ u16  hB[NB_ * D_];    // 16 KB, hidden activations bf16

    const int k = blockIdx.x;
    const int tid = threadIdx.x;
    const int wv = tid >> 5, lane = tid & 31;

    // y0 = init_mu + 0.05 * init_noise
    for (int idx = tid; idx < NB_ * D_; idx += 256) {
        int m = idx >> 9, d = idx & 511;
        float v = init_mu[k * D_ + d] + 0.05f * init_noise[((size_t)k * NB_ + m) * D_ + d];
        yF[idx] = v;
        yB[idx] = f32_to_bf16(v);
    }
    __syncthreads();

    const float* b1f = bf1 + (size_t)k * H_;
    const float* b2f = bf2 + (size_t)k * D_;
    const float* b1g = bg1 + (size_t)k * H_;
    const float* b2g = bg2 + (size_t)k * D_;

    for (int step = 0; step < STEPS_; ++step) {
        // Launder the k-index (NOT the pointers) through an opaque asm each
        // step: the weight addresses then cannot be proven loop-invariant, so
        // the fragment loads stay inside the step loop (avoiding the massive
        // hoist+spill-to-scratch), while the address arithmetic still roots at
        // the kernel-arg pointer Wt so address-space inference keeps the loads
        // as global_load_b128 (not flat_load, which would also tie up DScnt).
        unsigned kq = (unsigned)k;
        asm volatile("" : "+s"(kq));
        const u16* W1p  = Wt + ((size_t)(0 * 64 + kq)) * (D_ * H_);
        const u16* W2p  = Wt + ((size_t)(1 * 64 + kq)) * (D_ * H_);
        const u16* Wg1p = Wt + ((size_t)(2 * 64 + kq)) * (D_ * H_);
        const u16* Wg2p = Wt + ((size_t)(3 * 64 + kq)) * (D_ * H_);

        v8f tmp[4], accF[4], accG[4];

        // drift layer 1: h = tanh(y @ Wf1 + bf1)
#pragma unroll
        for (int nt = 0; nt < 4; ++nt) tmp[nt] = vzero8();
        wave_gemm4(yB, W1p, wv, lane, tmp);
        act_tanh_store(tmp, b1f, hB, wv, lane);
        __syncthreads();

        // drift layer 2: accF = h @ Wf2
#pragma unroll
        for (int nt = 0; nt < 4; ++nt) accF[nt] = vzero8();
        wave_gemm4(hB, W2p, wv, lane, accF);
        __syncthreads();   // everyone done reading hB

        // diffusion layer 1: h = tanh(y @ Wg1 + bg1)
#pragma unroll
        for (int nt = 0; nt < 4; ++nt) tmp[nt] = vzero8();
        wave_gemm4(yB, Wg1p, wv, lane, tmp);
        act_tanh_store(tmp, b1g, hB, wv, lane);
        __syncthreads();

        // diffusion layer 2: accG = h @ Wg2
#pragma unroll
        for (int nt = 0; nt < 4; ++nt) accG[nt] = vzero8();
        wave_gemm4(hB, Wg2p, wv, lane, accG);

        // Euler–Maruyama update (each lane owns disjoint (m,col) elements)
        const int saveT = step >> 2;
        const bool doSave = ((step & 3) == 3) && (saveT < T_);
#pragma unroll
        for (int nt = 0; nt < 4; ++nt) {
            int col = ((wv * 4 + nt) * 16) + (lane & 15);
            float bd = b2f[col];
            float bg = b2g[col];
#pragma unroll
            for (int v = 0; v < 8; ++v) {
                int m = v + ((lane >> 4) << 3);
                float dr = accF[nt][v] + bd;
                float gx = accG[nt][v] + bg;
                float gs = 0.1f / (1.0f + expf(-gx));
                float dwv = dW[(((size_t)step * K_ + k) * NB_ + m) * D_ + col];
                float yn = yF[m * D_ + col] + dr * DT_ + gs * (SQDT_ * dwv);
                yF[m * D_ + col] = yn;
                yB[m * D_ + col] = f32_to_bf16(yn);
                if (doSave)
                    pathBuf[(((size_t)saveT * K_ + k) * NB_ + m) * D_ + col] = yn;
            }
        }
        __syncthreads();
    }
}

// ---------- kernel 3: generic f32 -> bf16 convert ----------
__global__ __launch_bounds__(256) void cvt_bf16_kernel(const float* __restrict__ in,
                                                       u16* __restrict__ out, int n) {
    int i = blockIdx.x * 256 + threadIdx.x;
    if (i < n) out[i] = f32_to_bf16(in[i]);
}

// ---------- kernel 4: z row norms ----------
__global__ __launch_bounds__(256) void znorm_kernel(const float* __restrict__ z,
                                                    float* __restrict__ zn) {
    int wid = blockIdx.x * 8 + (threadIdx.x >> 5);  // 8*512 = 4096 rows
    int lane = threadIdx.x & 31;
    const float* row = z + (size_t)wid * D_;
    float s = 0.f;
    for (int d = lane; d < D_; d += 32) { float v = row[d]; s += v * v; }
#pragma unroll
    for (int o = 16; o; o >>= 1) s += __shfl_xor(s, o, 32);
    if (!lane) zn[wid] = s;
}

// ---------- kernel 5: G[t][i][b] = zhat_i . z_b (WMMA, one wave per tile) ----------
__global__ __launch_bounds__(256) void gemm_g_kernel(const u16* __restrict__ zhb,
                                                     const u16* __restrict__ zb,
                                                     float* __restrict__ G) {
    int wid = blockIdx.x * 8 + (threadIdx.x >> 5);   // 8 * 64 * 32 = 16384 tiles
    int lane = threadIdx.x & 31;
    int bt = wid & 31;
    int it = (wid >> 5) & 63;
    int t  = wid >> 11;
    const u16* A = zhb + (size_t)t * 1024 * D_;
    const u16* B = zb  + (size_t)t * 512 * D_;
    const u16* arowp = A + (size_t)(it * 16 + (lane & 15)) * D_;
    const u16* browp = B + (size_t)(bt * 16 + (lane & 15)) * D_;
    v8f acc = vzero8();
#pragma unroll 4
    for (int kt = 0; kt < 16; ++kt) {
        int dbase = kt * 32;
        v16bf a = loadA16(arowp, dbase, lane);
        v16bf b = loadB16(browp, dbase, lane);
        acc = __builtin_amdgcn_wmma_f32_16x16x32_bf16(false, a, false, b, (short)0, acc,
                                                      false, false);
    }
#pragma unroll
    for (int v = 0; v < 8; ++v) {
        int i = it * 16 + v + ((lane >> 4) << 3);
        int b = bt * 16 + (lane & 15);
        G[((size_t)t * 1024 + i) * 512 + b] = acc[v];
    }
}

// ---------- kernel 6: zero accumulators ----------
__global__ void zero_acc_kernel(float* acc) {
    if (threadIdx.x < 16) acc[threadIdx.x] = 0.0f;
}

// ---------- kernel 7: nearest-z pairing + MSE (one wave per path row) ----------
__global__ __launch_bounds__(256) void pair_mse_kernel(const float* __restrict__ all_z,
                                                       const int* __restrict__ all_y,
                                                       const float* __restrict__ pathBuf,
                                                       const float* __restrict__ G,
                                                       const float* __restrict__ zn,
                                                       float* __restrict__ acc) {
    int wid = blockIdx.x * 8 + (threadIdx.x >> 5);  // 8 * 1024 = 8192 waves
    int lane = threadIdx.x & 31;
    int t = wid >> 10;
    int i = wid & 1023;
    int pk = i >> 4;  // path_y
    const float* Gi  = G  + ((size_t)t * 1024 + i) * 512;
    const int*   yt  = all_y + (size_t)t * B_;
    const float* znt = zn + (size_t)t * B_;
    float best = -3.4e38f;
    int bb = 0;
    for (int b = lane; b < B_; b += 32) {
        float s = 2.0f * Gi[b] - znt[b] + ((yt[b] == pk) ? 0.0f : -1.0e6f);
        if (s > best) { best = s; bb = b; }
    }
#pragma unroll
    for (int o = 16; o; o >>= 1) {
        float ob = __shfl_xor(best, o, 32);
        int obb = __shfl_xor(bb, o, 32);
        if (ob > best || (ob == best && obb < bb)) { best = ob; bb = obb; }
    }
    const float* zr = all_z   + ((size_t)t * B_ + bb) * D_;
    const float* zh = pathBuf + ((size_t)t * 1024 + i) * D_;
    float s = 0.f;
    for (int d = lane; d < D_; d += 32) {
        float df = zh[d] - zr[d];
        s += df * df;
    }
#pragma unroll
    for (int o = 16; o; o >>= 1) s += __shfl_xor(s, o, 32);
    if (!lane) atomicAdd(&acc[t], s);
}

// ---------- kernel 8: contrastive CE (one wave per (t,b)) ----------
__global__ __launch_bounds__(256) void ce_kernel(const int* __restrict__ all_y,
                                                 const float* __restrict__ G,
                                                 float* __restrict__ acc) {
    int wid = blockIdx.x * 8 + (threadIdx.x >> 5);  // 8 * 512 = 4096 waves
    int lane = threadIdx.x & 31;
    int t = wid >> 9;
    int b = wid & 511;
    int yk = all_y[(size_t)t * B_ + b];
    const float* Gt = G + (size_t)t * 1024 * 512;
    float pos = 0.f, alls = 0.f;
    for (int i = lane; i < 1024; i += 32) {
        float s = Gt[(size_t)i * 512 + b];
        s = (s < 10.0f) ? s : 0.0f;   // sim * (sim < 10)
        float e = expf(s);
        alls += e;
        if ((i >> 4) == yk) pos += e;
    }
#pragma unroll
    for (int o = 16; o; o >>= 1) {
        alls += __shfl_xor(alls, o, 32);
        pos  += __shfl_xor(pos,  o, 32);
    }
    if (!lane) atomicAdd(&acc[8 + t], logf(alls + EPS_) - logf(pos + EPS_));
}

// ---------- kernel 9: final reduction ----------
__global__ void final_kernel(const float* __restrict__ acc, float* __restrict__ out) {
    if (threadIdx.x == 0) {
        float s = 0.f;
        for (int t = 0; t < T_; ++t) {
            float mse = acc[t] / (1024.0f * 512.0f);
            float ce  = acc[8 + t] / 512.0f;
            s += ce + 1.0f * mse;  // PATH_W = 1.0
        }
        out[0] = s / (float)T_;
    }
}

// ---------- host launcher ----------
extern "C" void kernel_launch(void* const* d_in, const int* in_sizes, int n_in,
                              void* d_out, int out_size, void* d_ws, size_t ws_size,
                              hipStream_t stream) {
    const float* all_z      = (const float*)d_in[0];
    const int*   all_y      = (const int*)  d_in[1];
    const float* init_mu    = (const float*)d_in[2];
    const float* Wf1        = (const float*)d_in[3];
    const float* bf1        = (const float*)d_in[4];
    const float* Wf2        = (const float*)d_in[5];
    const float* bf2        = (const float*)d_in[6];
    const float* Wg1        = (const float*)d_in[7];
    const float* bg1        = (const float*)d_in[8];
    const float* Wg2        = (const float*)d_in[9];
    const float* bg2        = (const float*)d_in[10];
    const float* init_noise = (const float*)d_in[11];
    const float* dW         = (const float*)d_in[12];
    float* out = (float*)d_out;

    char* ws = (char*)d_ws;
    u16*   Wt      = (u16*)  (ws);
    float* pathBuf = (float*)(ws + WT_BYTES);
    u16*   zb      = (u16*)  (ws + WT_BYTES + PATH_BYTES);
    u16*   zhb     = (u16*)  (ws + WT_BYTES + PATH_BYTES + ZB_BYTES);
    float* G       = (float*)(ws + WT_BYTES + PATH_BYTES + ZB_BYTES + ZHB_BYTES);
    float* zn      = (float*)(ws + WT_BYTES + PATH_BYTES + ZB_BYTES + ZHB_BYTES + G_BYTES);
    float* acc     = (float*)(ws + WT_BYTES + PATH_BYTES + ZB_BYTES + ZHB_BYTES + G_BYTES + ZN_BYTES);

    // 1) pack weights to bf16, transposed to [h][d]
    pack_kernel<<<4 * 64 * 16 * 16, 256, 0, stream>>>(Wf1, Wf2, Wg1, Wg2, Wt);

    // 2) the SDE chain
    sde_kernel<<<K_, 256, 0, stream>>>(init_mu, init_noise, dW,
                                       bf1, bf2, bg1, bg2, Wt, pathBuf);

    // 3) convert z and zhat to bf16
    cvt_bf16_kernel<<<(T_ * B_ * D_) / 256, 256, 0, stream>>>(all_z, zb, T_ * B_ * D_);
    cvt_bf16_kernel<<<(T_ * 1024 * D_) / 256, 256, 0, stream>>>(pathBuf, zhb, T_ * 1024 * D_);

    // 4) z row norms
    znorm_kernel<<<(T_ * B_) / 8, 256, 0, stream>>>(all_z, zn);

    // 5) similarity GEMM G = zhat @ z^T (per t)
    gemm_g_kernel<<<(T_ * 64 * 32) / 8, 256, 0, stream>>>(zhb, zb, G);

    // 6) zero loss accumulators
    zero_acc_kernel<<<1, 32, 0, stream>>>(acc);

    // 7) nearest-pair + MSE
    pair_mse_kernel<<<(T_ * 1024) / 8, 256, 0, stream>>>(all_z, all_y, pathBuf, G, zn, acc);

    // 8) contrastive CE
    ce_kernel<<<(T_ * B_) / 8, 256, 0, stream>>>(all_y, G, acc);

    // 9) final mean
    final_kernel<<<1, 64, 0, stream>>>(acc, out);
}
